// RecurrentLRSCell_54365696032793
// MI455X (gfx1250) — matrix operand order repaired and checked by reading, not compiled, not run on hardware
//
#include <hip/hip_runtime.h>
#include <math.h>

typedef __attribute__((ext_vector_type(16))) _Float16 v16h;
typedef __attribute__((ext_vector_type(8)))  _Float16 v8h;
typedef __attribute__((ext_vector_type(8)))  float    v8f;
typedef __attribute__((ext_vector_type(4)))  float    v4f;

#define B_     256
#define T_     1024
#define F_     128
#define D_     256
#define UNITS_ 64

static __device__ inline v16h cat8(v8h lo, v8h hi) {
    return __builtin_shufflevector(lo, hi, 0, 1, 2, 3, 4, 5, 6, 7,
                                           8, 9, 10, 11, 12, 13, 14, 15);
}

// Issue one 16-byte async global->LDS copy (CDNA5 async path, ASYNCcnt).
static __device__ inline void async_ld16(const void* gsrc, void* ldsDst) {
    unsigned lo = (unsigned)(uintptr_t)ldsDst;  // generic LDS addr: low 32 = LDS offset
    asm volatile("global_load_async_to_lds_b128 %0, %1, off"
                 :: "v"(lo), "v"(gsrc) : "memory");
}

// ---------------------------------------------------------------------------
// Kernel 0: transpose + f32->f16 convert the two weight matrices into d_ws.
//   KT[n][f] = kernel[f][n]            (D_ x F_  halfs)
//   RT[n][k] = recurrent_kernel[k][n]  (D_ x D_  halfs)
// ---------------------------------------------------------------------------
__global__ __launch_bounds__(256) void prep_weights(
    const float* __restrict__ Kf, const float* __restrict__ Rf,
    _Float16* __restrict__ KT, _Float16* __restrict__ RT) {
    int i = blockIdx.x * blockDim.x + threadIdx.x;
    if (i < D_ * F_) {
        int n = i / F_, f = i % F_;
        KT[i] = (_Float16)Kf[f * D_ + n];
    } else if (i < D_ * F_ + D_ * D_) {
        int j = i - D_ * F_;
        int n = j / D_, k = j % D_;
        RT[j] = (_Float16)Rf[k * D_ + n];
    }
}

// ---------------------------------------------------------------------------
// Kernel 1: U[b,t,:] = (x[b,t,:] - x[b,t-1,:]) @ kernel + bias.
// Grid: (B*T/16), block 128 (4 waves). dx staged once in LDS (f16), each wave
// owns 4 N-tiles -> 16 WMMAs/wave; x is read exactly twice per element.
// ---------------------------------------------------------------------------
__global__ __launch_bounds__(128) void xproj_kernel(
    const float* __restrict__ x, const _Float16* __restrict__ KT,
    const float* __restrict__ bias, float* __restrict__ U) {
    __shared__ _Float16 dx16[16 * F_];  // 4 KB

    const int tid  = threadIdx.x;
    const int lane = tid & 31;
    const int wave = tid >> 5;
    const int half = lane >> 4;
    const int l16  = lane & 15;
    const int mBase = blockIdx.x * 16;

    // Stage dx (f16) into LDS: 2048 elems, 16 per thread.
    {
        const int row = tid >> 3;
        const int seg = (tid & 7) * 16;
        const int r   = mBase + row;
        const bool hasPrev = (r & (T_ - 1)) != 0;
        const float* xr = x + (size_t)r * F_ + seg;
        v4f a[4];
#pragma unroll
        for (int c = 0; c < 4; ++c) a[c] = *(const v4f*)(xr + c * 4);
        if (hasPrev) {
            const float* xp = xr - F_;
#pragma unroll
            for (int c = 0; c < 4; ++c) {
                v4f p = *(const v4f*)(xp + c * 4);
#pragma unroll
                for (int i = 0; i < 4; ++i) a[c][i] -= p[i];
            }
        }
        _Float16* d = dx16 + row * F_ + seg;
#pragma unroll
        for (int c = 0; c < 4; ++c)
#pragma unroll
            for (int i = 0; i < 4; ++i) d[c * 4 + i] = (_Float16)a[c][i];
    }
    __syncthreads();

    const int nBase = wave * 64;  // this wave's 4 N tiles
    v8f acc[4];
#pragma unroll
    for (int j = 0; j < 4; ++j) {
        const float bv = bias[nBase + j * 16 + l16];
#pragma unroll
        for (int i = 0; i < 8; ++i) acc[j][i] = bv;
    }

    const _Float16* ap = dx16 + l16 * F_;
#pragma unroll
    for (int kk = 0; kk < 4; ++kk) {
        const int k0 = kk * 32;
        v16h a = cat8(*(const v8h*)(ap + k0 + half * 8),
                      *(const v8h*)(ap + k0 + 16 + half * 8));
#pragma unroll
        for (int j = 0; j < 4; ++j) {
            const _Float16* bp = KT + (size_t)(nBase + j * 16 + l16) * F_;
            v16h b = cat8(*(const v8h*)(bp + k0 + half * 8),
                          *(const v8h*)(bp + k0 + 16 + half * 8));
            acc[j] = __builtin_amdgcn_wmma_f32_16x16x32_f16(
                false, a, false, b, (short)0, acc[j], false, false);
        }
    }

#pragma unroll
    for (int j = 0; j < 4; ++j)
#pragma unroll
        for (int v = 0; v < 8; ++v) {
            const int m = v + half * 8;
            U[(size_t)(mBase + m) * D_ + nBase + j * 16 + l16] = acc[j][v];
        }
}

// ---------------------------------------------------------------------------
// Kernel 2: sequential scan. One WG per 16 batch rows; R (f16) LDS-resident;
// U[t+1] async-double-buffered into LDS while step t computes.
// LDS: Rl 128K | dh16 8K | h x3 48K | Ubuf x2 32K = 216K (CDNA5 320K WGP LDS)
// ---------------------------------------------------------------------------
#define RL_OFF 0
#define DH_OFF (D_ * D_ * 2)                 // 131072
#define HB_OFF (DH_OFF + 16 * D_ * 2)        // 139264
#define UB_OFF (HB_OFF + 3 * 16 * D_ * 4)    // 188416
#define LDS_TOTAL (UB_OFF + 2 * 16 * D_ * 4) // 221184

static __device__ inline void issue_u_tile(const float* __restrict__ HU,
                                           float* ubuf, int bBase, int tn, int tid) {
    const int rowm = tid >> 6;       // 0..3
    const int cc   = tid & 63;       // 16B chunk within row
    float* dstBase = ubuf + (tn & 1) * (16 * D_);
#pragma unroll
    for (int q = 0; q < 4; ++q) {
        const int m = rowm + q * 4;
        const float* gsrc = HU + ((size_t)(bBase + m) * T_ + tn) * D_ + cc * 4;
        async_ld16(gsrc, dstBase + m * D_ + cc * 4);
    }
}

__global__ __launch_bounds__(256) void recurrence_kernel(
    const _Float16* __restrict__ RT, float* __restrict__ HU) {
    extern __shared__ char smem[];
    _Float16* Rl   = (_Float16*)(smem + RL_OFF);
    _Float16* dh16 = (_Float16*)(smem + DH_OFF);
    float*    hb   = (float*)(smem + HB_OFF);
    float*    ubuf = (float*)(smem + UB_OFF);

    const int tid  = threadIdx.x;
    const int lane = tid & 31;
    const int wave = tid >> 5;
    const int half = lane >> 4;
    const int l16  = lane & 15;
    const int bBase = blockIdx.x * 16;

    // Stage RT (128KB) into LDS via the async engine.
    {
        const char* src = (const char*)RT;
        char* dst = (char*)Rl;
        for (int i = tid; i < (D_ * D_ * 2) / 16; i += 256)
            async_ld16(src + i * 16, dst + i * 16);
    }
    // Zero h1/h2 while the copies are in flight.
    for (int i = tid; i < 2 * 16 * D_; i += 256) hb[i] = 0.0f;
    asm volatile("s_wait_asynccnt 0x0" ::: "memory");

    // Prefetch U for t = 0.
    issue_u_tile(HU, ubuf, bBase, 0, tid);
    __syncthreads();

    int i1 = 0, i2 = 1, iw = 2;

    const int n0 = wave * 32;  // this wave's two N tiles: n0, n0+16
    const _Float16* bp0 = Rl + (size_t)(n0 + l16) * D_;
    const _Float16* bp1 = Rl + (size_t)(n0 + 16 + l16) * D_;
    const _Float16* ap  = dh16 + l16 * D_;

    for (int t = 0; t < T_; ++t) {
        const float* h1 = hb + i1 * 16 * D_;
        const float* h2 = hb + i2 * 16 * D_;
        float*       hw = hb + iw * 16 * D_;
        const float* ub = ubuf + (t & 1) * (16 * D_);

        // Kick off next step's U transfer, overlapping with this step's math.
        if (t + 1 < T_) issue_u_tile(HU, ubuf, bBase, t + 1, tid);

        // dh16 = f16(h1 - h2), one cooperative pass (16 elems/thread).
        {
            const int base = tid * 16;
#pragma unroll
            for (int c = 0; c < 4; ++c) {
                v4f a = *(const v4f*)(h1 + base + c * 4);
                v4f b = *(const v4f*)(h2 + base + c * 4);
                _Float16* d = dh16 + base + c * 4;
#pragma unroll
                for (int i = 0; i < 4; ++i) d[i] = (_Float16)(a[i] - b[i]);
            }
        }
        // Ensure this step's U chunks (issued last iteration) have landed:
        // per-wave ASYNCcnt is in-order, the newest 4 are for t+1.
        if (t + 1 < T_) asm volatile("s_wait_asynccnt 0x4" ::: "memory");
        else            asm volatile("s_wait_asynccnt 0x0" ::: "memory");
        __syncthreads();

        v8f acc[2];
#pragma unroll
        for (int j = 0; j < 2; ++j)
#pragma unroll
            for (int i = 0; i < 8; ++i) acc[j][i] = 0.0f;

#pragma unroll
        for (int kk = 0; kk < 8; ++kk) {
            const int k0 = kk * 32;
            v16h a = cat8(*(const v8h*)(ap + k0 + half * 8),
                          *(const v8h*)(ap + k0 + 16 + half * 8));
            v16h b0 = cat8(*(const v8h*)(bp0 + k0 + half * 8),
                           *(const v8h*)(bp0 + k0 + 16 + half * 8));
            acc[0] = __builtin_amdgcn_wmma_f32_16x16x32_f16(
                false, a, false, b0, (short)0, acc[0], false, false);
            v16h b1 = cat8(*(const v8h*)(bp1 + k0 + half * 8),
                           *(const v8h*)(bp1 + k0 + 16 + half * 8));
            acc[1] = __builtin_amdgcn_wmma_f32_16x16x32_f16(
                false, a, false, b1, (short)0, acc[1], false, false);
        }

        // u = U + dh@R; ht = tanh(h1 + shifted*u); write global + LDS h.
#pragma unroll
        for (int j = 0; j < 2; ++j) {
            const int n = n0 + j * 16 + l16;
#pragma unroll
            for (int v = 0; v < 8; ++v) {
                const int m = v + half * 8;
                const float u   = ub[m * D_ + n] + acc[j][v];
                const float h1v = h1[m * D_ + n];
                const float sh  = (n < UNITS_) ? 1.0f : h1[m * D_ + n - UNITS_];
                const float ht  = tanhf(h1v + sh * u);
                HU[((size_t)(bBase + m) * T_ + t) * D_ + n] = ht;
                hw[m * D_ + n] = ht;
            }
        }
        __syncthreads();
        const int tmp = i2; i2 = i1; i1 = iw; iw = tmp;
    }
}

// ---------------------------------------------------------------------------
extern "C" void kernel_launch(void* const* d_in, const int* in_sizes, int n_in,
                              void* d_out, int out_size, void* d_ws, size_t ws_size,
                              hipStream_t stream) {
    (void)in_sizes; (void)n_in; (void)out_size; (void)ws_size;
    const float* x    = (const float*)d_in[0];
    const float* Kf   = (const float*)d_in[1];
    const float* Rf   = (const float*)d_in[2];
    const float* bias = (const float*)d_in[3];
    float* out = (float*)d_out;

    _Float16* KT = (_Float16*)d_ws;                         // D_*F_ halfs
    _Float16* RT = (_Float16*)((char*)d_ws + D_ * F_ * 2);  // D_*D_ halfs

    prep_weights<<<(D_ * F_ + D_ * D_ + 255) / 256, 256, 0, stream>>>(Kf, Rf, KT, RT);

    xproj_kernel<<<B_ * T_ / 16, 128, 0, stream>>>(x, KT, bias, out);

    recurrence_kernel<<<B_ / 16, 256, (size_t)LDS_TOTAL, stream>>>(RT, out);
}
